// FpgnnModel_10797547782760
// MI455X (gfx1250) — matrix operand order repaired and checked by reading, compiled, not run on hardware
//
#include <hip/hip_runtime.h>

typedef __bf16 bf16_t;
typedef __attribute__((ext_vector_type(16))) bf16_t v16bf;
typedef __attribute__((ext_vector_type(2)))  bf16_t v2bf;
typedef __attribute__((ext_vector_type(8)))  float  v8f;
typedef __attribute__((ext_vector_type(4)))  unsigned u32x4;

// ---------- helpers ----------
__device__ __forceinline__ unsigned short bf16bits(float f) {
  unsigned u = __builtin_bit_cast(unsigned, f);
  return (unsigned short)((u + 0x7FFFu + ((u >> 16) & 1u)) >> 16);   // RNE
}
__device__ __forceinline__ bf16_t f2bf(float f) {
  return __builtin_bit_cast(bf16_t, bf16bits(f));
}
__device__ __forceinline__ float bf2float(bf16_t b) {
  unsigned u = ((unsigned)__builtin_bit_cast(unsigned short, b)) << 16;
  return __builtin_bit_cast(float, u);
}
#if __has_builtin(__builtin_amdgcn_cvt_pk_bf16_f32)
__device__ __forceinline__ unsigned pack2bf(float a, float b) {
  v2bf p = __builtin_amdgcn_cvt_pk_bf16_f32(a, b);   // v_cvt_pk_bf16_f32
  return __builtin_bit_cast(unsigned, p);
}
#else
__device__ __forceinline__ unsigned pack2bf(float a, float b) {
  return (unsigned)bf16bits(a) | ((unsigned)bf16bits(b) << 16);
}
#endif

// Fragment-order LDS tiles: per 16x32 A-tile / 32x16 B-tile, layout is
// [lane(32)][elem(16)] bf16 (1KB). A lane's 16 values are contiguous ->
// one 32B vector load (2x ds_load_b128).
//
// A swizzle (contiguous k order, not the ISA table order; 16-bit A 16x32): element (m,kk) ->
//   lane = m + 16*(kk>>4),     idx = kk & 15
// B swizzle (32x16 KxN):       element (kk,n) ->
//   lane = n + 16*(kk>>4),     idx = kk & 15
__device__ __forceinline__ v16bf load_frag(const bf16_t* tile, int lane) {
  return *(const v16bf*)(tile + lane * 16);
}
__device__ __forceinline__ v8f wmma_bf16(v16bf a, v16bf b, v8f c) {
  return __builtin_amdgcn_wmma_f32_16x16x32_bf16(false, a, false, b, (short)0, c, false, false);
}

__device__ __forceinline__ float leaky02(float x) { return x >= 0.f ? x : 0.2f * x; }
__device__ __forceinline__ float eluf(float x)    { return x > 0.f ? x : (__expf(x) - 1.0f); }

#define NEG_BIG (-9e15f)

// ---------------------------------------------------------------------------
// Kernel 1: fused GAT layer 1 per (head, batch).
// ---------------------------------------------------------------------------
__global__ __launch_bounds__(256) void gat1_fused(
    const float* __restrict__ h, const int* __restrict__ adj,
    const float* __restrict__ edge, const float* __restrict__ Whead,
    const float* __restrict__ a1h, const float* __restrict__ a2h,
    const float* __restrict__ aeh, bf16_t* __restrict__ xB)
{
  __shared__ alignas(32) char smem[111616];
  bf16_t* hA    = (bf16_t*)smem;             // A-frag tiles [mtile8][kc5][32][16] = 40KB
                                             // reused as attA [mtile8][kc4][32][16] = 32KB
  bf16_t* WsB   = (bf16_t*)(smem + 40960);   // B-frag tiles [kc5*4+nt][32][16] = 20KB
  float*  WhF   = (float*)(smem + 61440);    // [128][64] f32 = 32KB
  bf16_t* WhBsw = (bf16_t*)(smem + 94208);   // B-frag tiles [kc4*4+nt][32][16] = 16KB
  float*  s1s   = (float*)(smem + 110592);   // [128]
  float*  s2s   = s1s + 128;                 // [128]

  const int hh = blockIdx.x;                 // head
  const int b  = blockIdx.y;                 // batch
  const int tid = threadIdx.x;
  const int lane = tid & 31;
  const int wv = tid >> 5;

  // ---- stage h_b as A-fragments (K padded to 160) ----
  const float* hsrc = h + (size_t)b * 128 * 133;
  for (int s = tid; s < 2560; s += 256) {          // 8mt*5kc*32ln*2grp
    int grp = s & 1, ln = (s >> 1) & 31, tk = s >> 6;   // tk = mtile*5+kc
    int kc = tk % 5, mtile = tk / 5;
    int m = ln & 15, hi = ln >> 4;
    int kb = kc * 32 + hi * 16 + grp * 8;
    const float* src = hsrc + (mtile * 16 + m) * 133;
    float v[8];
#pragma unroll
    for (int e = 0; e < 8; ++e) v[e] = (kb + e < 133) ? src[kb + e] : 0.f;
    u32x4 pk = { pack2bf(v[0], v[1]), pack2bf(v[2], v[3]),
                 pack2bf(v[4], v[5]), pack2bf(v[6], v[7]) };
    *(u32x4*)(hA + (size_t)tk * 512 + ln * 16 + grp * 8) = pk;
  }
  // ---- stage W_h as B-fragments ----
  const float* wsrc = Whead + (size_t)hh * 133 * 64;
  for (int s = tid; s < 640; s += 256) {           // 5kc*4nt*32ln
    int ln = s & 31, tile = s >> 5;                // tile = kc*4+ntile
    int kc = tile >> 2, ntile = tile & 3;
    int n = ntile * 16 + (ln & 15);
    int kb = kc * 32 + ((ln >> 4) << 4);
    unsigned pk[8];
#pragma unroll
    for (int p = 0; p < 8; ++p) {
      int k0 = kb + 2 * p;
      float f0 = (k0 < 133) ? wsrc[k0 * 64 + n] : 0.f;
      float f1 = (k0 + 1 < 133) ? wsrc[(k0 + 1) * 64 + n] : 0.f;
      pk[p] = pack2bf(f0, f1);
    }
    u32x4* dst = (u32x4*)(WsB + (size_t)tile * 512 + ln * 16);
    u32x4 lo = { pk[0], pk[1], pk[2], pk[3] };
    u32x4 hi4 = { pk[4], pk[5], pk[6], pk[7] };
    dst[0] = lo; dst[1] = hi4;
  }
  __syncthreads();

  // ---- GEMM1: Wh = h @ W ----
  const int m0 = wv * 16;
  v8f acc[4] = {};
#pragma unroll
  for (int kc = 0; kc < 5; ++kc) {
    v16bf af = load_frag(hA + (size_t)(wv * 5 + kc) * 512, lane);
#pragma unroll
    for (int t = 0; t < 4; ++t) {
      v16bf bfv = load_frag(WsB + (size_t)(kc * 4 + t) * 512, lane);
      acc[t] = wmma_bf16(af, bfv, acc[t]);
    }
  }
  const int half8 = (lane >> 4) * 8;
  const int nlo = lane & 15;
#pragma unroll
  for (int t = 0; t < 4; ++t)
#pragma unroll
    for (int r = 0; r < 8; ++r) {
      int mrow = m0 + half8 + r, n = t * 16 + nlo;
      float v = acc[t][r];
      WhF[mrow * 64 + n] = v;
      // scatter into B-fragment order for GEMM2 (K=128 rows, N=64 cols)
      int kc2 = mrow >> 5, kk = mrow & 31;
      int lane2 = nlo + ((kk >> 4) << 4);
      WhBsw[(size_t)((kc2 * 4 + t) * 32 + lane2) * 16 + (kk & 15)] = f2bf(v);
    }
  __syncthreads();

  // ---- s1 = Wh . a1, s2 = Wh . a2 ----
  if (tid < 128) {
    const float* a1 = a1h + hh * 64;
    const float* a2 = a2h + hh * 64;
    float d1 = 0.f, d2 = 0.f;
    for (int d = 0; d < 64; ++d) {
      float w = WhF[tid * 64 + d];
      d1 += w * a1[d];
      d2 += w * a2[d];
    }
    s1s[tid] = d1; s2s[tid] = d2;
  }
  __syncthreads();

  // ---- attention + row softmax, written in A-fragment order ----
  bf16_t* attA = hA;
  const float ae = aeh[hh];
  const size_t ebase = (size_t)b * 16384;
  const int ahi = (lane >> 4) << 4;     // kk = lane
  const int aidx = lane & 15;
  for (int r = 0; r < 16; ++r) {
    int i = wv * 16 + r;
    float vals[4];
#pragma unroll
    for (int c = 0; c < 4; ++c) {
      int j = lane + 32 * c;
      float vv = leaky02(s1s[i] + s2s[j] + edge[ebase + i * 128 + j] * ae);
      vals[c] = (adj[ebase + i * 128 + j] > 0) ? vv : NEG_BIG;
    }
    float mx = fmaxf(fmaxf(vals[0], vals[1]), fmaxf(vals[2], vals[3]));
    for (int s = 16; s >= 1; s >>= 1) mx = fmaxf(mx, __shfl_xor(mx, s));
    float p[4], sum = 0.f;
#pragma unroll
    for (int c = 0; c < 4; ++c) { p[c] = __expf(vals[c] - mx); sum += p[c]; }
    for (int s = 16; s >= 1; s >>= 1) sum += __shfl_xor(sum, s);
    float inv = 1.0f / sum;
    int mtile = i >> 4, m = i & 15;
#pragma unroll
    for (int c = 0; c < 4; ++c)
      attA[(size_t)((mtile * 4 + c) * 32 + m + ahi) * 16 + aidx] = f2bf(p[c] * inv);
  }
  __syncthreads();

  // ---- GEMM2: hp = att @ Wh, ELU, store x[b, n, hh*64+d] ----
  v8f acc2[4] = {};
#pragma unroll
  for (int kc = 0; kc < 4; ++kc) {
    v16bf af = load_frag(attA + (size_t)(wv * 4 + kc) * 512, lane);
#pragma unroll
    for (int t = 0; t < 4; ++t) {
      v16bf bfv = load_frag(WhBsw + (size_t)(kc * 4 + t) * 512, lane);
      acc2[t] = wmma_bf16(af, bfv, acc2[t]);
    }
  }
#pragma unroll
  for (int t = 0; t < 4; ++t)
#pragma unroll
    for (int r = 0; r < 8; ++r) {
      int mrow = m0 + half8 + r, n = t * 16 + nlo;
      float v = eluf(acc2[t][r]);
      xB[((size_t)b * 128 + mrow) * 512 + hh * 64 + n] = f2bf(v);
    }
}

// ---------------------------------------------------------------------------
// Kernel 2: Wx = x @ W_out  (65536x512 @ 512x512) -> bf16 Wx, fused t1/t2 dots.
// A operand staged with GLOBAL_LOAD_ASYNC_TO_LDS_B128 (per-lane global gather,
// per-lane LDS scatter -> fragment swizzle for free), double-buffered.
// Uses only dynamic LDS so async LDS byte offsets start at 0.
// ---------------------------------------------------------------------------
extern __shared__ char dynsm[];

__global__ __launch_bounds__(256) void gemm_wout(
    const bf16_t* __restrict__ xB, const float* __restrict__ Wout,
    const float* __restrict__ a1o, const float* __restrict__ a2o,
    bf16_t* __restrict__ WxB, float* __restrict__ t1, float* __restrict__ t2)
{
  // layout (bytes): A bufs [2][8192] at 0, B bufs [2][4096] at 16384
  bf16_t* As = (bf16_t*)dynsm;
  bf16_t* Bs = (bf16_t*)(dynsm + 16384);
  const int tid = threadIdx.x, lane = tid & 31, wv = tid >> 5;
  const size_t m0g = (size_t)blockIdx.x * 128;
  const int n0g = blockIdx.y * 64;
  const unsigned long long xbase = (unsigned long long)(uintptr_t)xB;

  // async A staging: each lane copies one 16B fragment segment
  auto stageA = [&](int kc, int buf) {
#pragma unroll
    for (int s0 = 0; s0 < 2; ++s0) {
      int s = tid + s0 * 256;                       // 512 segments
      int grp = s & 1, ln = (s >> 1) & 31, mtile = s >> 6;
      int m = ln & 15, hi = ln >> 4;
      unsigned row = (unsigned)m0g + mtile * 16 + m;
      unsigned goff = row * 1024u + (unsigned)(kc * 32 + hi * 16 + grp * 8) * 2u;
      unsigned loff = (unsigned)(buf * 8192 + mtile * 1024 + ln * 32 + grp * 16);
      asm volatile("global_load_async_to_lds_b128 %0, %1, %2"
                   :: "v"(loff), "v"(goff), "s"(xbase) : "memory");
    }
  };
  // B staging: f32 -> bf16 conversion through VGPRs
  auto stageB = [&](int kc, int buf) {
    for (int s = tid; s < 128; s += 256) {
      int ln = s & 31, ntile = s >> 5;
      int gn = n0g + ntile * 16 + (ln & 15);
      int kb = kc * 32 + ((ln >> 4) << 4);
      unsigned pk[8];
#pragma unroll
      for (int p = 0; p < 8; ++p)
        pk[p] = pack2bf(Wout[(size_t)(kb + 2 * p) * 512 + gn],
                        Wout[(size_t)(kb + 2 * p + 1) * 512 + gn]);
      u32x4* dst = (u32x4*)(Bs + (size_t)(buf * 4 + ntile) * 512 + ln * 16);
      u32x4 lo = { pk[0], pk[1], pk[2], pk[3] };
      u32x4 hi4 = { pk[4], pk[5], pk[6], pk[7] };
      dst[0] = lo; dst[1] = hi4;
    }
  };

  v8f acc[4] = {};
  stageA(0, 0);
  stageB(0, 0);
  asm volatile("s_wait_asynccnt 0" ::: "memory");
  __syncthreads();
  for (int kc = 0; kc < 16; ++kc) {
    int buf = kc & 1;
    if (kc < 15) { stageA(kc + 1, buf ^ 1); stageB(kc + 1, buf ^ 1); }
    v16bf af = load_frag(As + (size_t)(buf * 8 + wv) * 512, lane);
#pragma unroll
    for (int t = 0; t < 4; ++t) {
      v16bf bfv = load_frag(Bs + (size_t)(buf * 4 + t) * 512, lane);
      acc[t] = wmma_bf16(af, bfv, acc[t]);
    }
    asm volatile("s_wait_asynccnt 0" ::: "memory");
    __syncthreads();
  }
  const int half8 = (lane >> 4) * 8, nlo = lane & 15;
#pragma unroll
  for (int r = 0; r < 8; ++r) {
    size_t m = m0g + wv * 16 + half8 + r;
    float p1 = 0.f, p2 = 0.f;
#pragma unroll
    for (int t = 0; t < 4; ++t) {
      int ng = n0g + t * 16 + nlo;
      float v = acc[t][r];
      WxB[m * 512 + ng] = f2bf(v);
      p1 += v * a1o[ng];
      p2 += v * a2o[ng];
    }
    p1 += __shfl_xor(p1, 1); p1 += __shfl_xor(p1, 2); p1 += __shfl_xor(p1, 4); p1 += __shfl_xor(p1, 8);
    p2 += __shfl_xor(p2, 1); p2 += __shfl_xor(p2, 2); p2 += __shfl_xor(p2, 4); p2 += __shfl_xor(p2, 8);
    if (nlo == 0) { atomicAdd(&t1[m], p1); atomicAdd(&t2[m], p2); }
  }
}

// ---------------------------------------------------------------------------
// Kernel 3: GAT layer 2 per batch: softmax, hp2 = att2 @ Wx, ELU, node-mean
// ---------------------------------------------------------------------------
__global__ __launch_bounds__(256) void gat2_fused(
    const int* __restrict__ adj, const float* __restrict__ edge,
    const float* __restrict__ aeo, const float* __restrict__ t1,
    const float* __restrict__ t2, const bf16_t* __restrict__ WxB,
    float* __restrict__ gat_out)
{
  __shared__ alignas(32) bf16_t attA[32 * 512];  // A-frag tiles [mtile8][kc4][32][16]
  __shared__ alignas(32) bf16_t Bs[8 * 512];     // B-frag tiles [ntile8][32][16]
  __shared__ float t1s[128], t2s[128];
  __shared__ float colsum[128];
  const int b = blockIdx.x, tid = threadIdx.x, lane = tid & 31, wv = tid >> 5;
  if (tid < 128) { t1s[tid] = t1[b * 128 + tid]; t2s[tid] = t2[b * 128 + tid]; }
  __syncthreads();

  const float ae = aeo[0];
  const size_t ebase = (size_t)b * 16384;
  const int ahi = (lane >> 4) << 4;
  const int aidx = lane & 15;
  for (int r = 0; r < 16; ++r) {
    int i = wv * 16 + r;
    float vals[4];
#pragma unroll
    for (int c = 0; c < 4; ++c) {
      int j = lane + 32 * c;
      float vv = leaky02(t1s[i] + t2s[j] + edge[ebase + i * 128 + j] * ae);
      vals[c] = (adj[ebase + i * 128 + j] > 0) ? vv : NEG_BIG;
    }
    float mx = fmaxf(fmaxf(vals[0], vals[1]), fmaxf(vals[2], vals[3]));
    for (int s = 16; s >= 1; s >>= 1) mx = fmaxf(mx, __shfl_xor(mx, s));
    float p[4], sum = 0.f;
#pragma unroll
    for (int c = 0; c < 4; ++c) { p[c] = __expf(vals[c] - mx); sum += p[c]; }
    for (int s = 16; s >= 1; s >>= 1) sum += __shfl_xor(sum, s);
    float inv = 1.0f / sum;
    int mtile = i >> 4, m = i & 15;
#pragma unroll
    for (int c = 0; c < 4; ++c)
      attA[(size_t)((mtile * 4 + c) * 32 + m + ahi) * 16 + aidx] = f2bf(p[c] * inv);
  }
  __syncthreads();

  const int nlo = lane & 15;
  for (int np = 0; np < 4; ++np) {            // 512 cols in 4 passes of 128
    if (tid < 128) colsum[tid] = 0.f;
    v8f acc[8] = {};
    __syncthreads();
    for (int kc = 0; kc < 4; ++kc) {
      {
        int s = tid;                           // exactly 256 segments
        int ln = s & 31, ntile = s >> 5;
        int gn = np * 128 + ntile * 16 + (ln & 15);
        int kb = kc * 32 + ((ln >> 4) << 4);
        const bf16_t* src = WxB + ((size_t)b * 128 + kb) * 512 + gn;
        unsigned pk[8];
#pragma unroll
        for (int p = 0; p < 8; ++p) {
          unsigned e0 = __builtin_bit_cast(unsigned short, src[(size_t)(2 * p) * 512]);
          unsigned e1 = __builtin_bit_cast(unsigned short, src[(size_t)(2 * p + 1) * 512]);
          pk[p] = e0 | (e1 << 16);
        }
        u32x4* dst = (u32x4*)(Bs + (size_t)ntile * 512 + ln * 16);
        u32x4 lo = { pk[0], pk[1], pk[2], pk[3] };
        u32x4 hi4 = { pk[4], pk[5], pk[6], pk[7] };
        dst[0] = lo; dst[1] = hi4;
      }
      __syncthreads();
      v16bf af = load_frag(attA + (size_t)(wv * 4 + kc) * 512, lane);
#pragma unroll
      for (int t = 0; t < 8; ++t) {
        v16bf bfv = load_frag(Bs + (size_t)t * 512, lane);
        acc[t] = wmma_bf16(af, bfv, acc[t]);
      }
      __syncthreads();
    }
#pragma unroll
    for (int t = 0; t < 8; ++t) {
      float part = 0.f;
#pragma unroll
      for (int r = 0; r < 8; ++r) part += eluf(acc[t][r]);
      part += __shfl_xor(part, 16);
      if (lane < 16) atomicAdd(&colsum[t * 16 + nlo], part);
    }
    __syncthreads();
    if (tid < 128) gat_out[(size_t)b * 512 + np * 128 + tid] = colsum[tid] * (1.0f / 128.0f);
    __syncthreads();
  }
}

// ---------------------------------------------------------------------------
// Generic bounds-guarded WMMA GEMM: out = act(A @ B + bias + addend)
// ---------------------------------------------------------------------------
__global__ __launch_bounds__(256) void gemm_generic(
    const void* __restrict__ Aptr, int a_bf16,
    const float* __restrict__ Bw, const float* __restrict__ bias,
    const float* __restrict__ addend,
    float* __restrict__ outF, bf16_t* __restrict__ outB,
    int M, int N, int K, int act)
{
  __shared__ alignas(32) bf16_t As[8 * 512];
  __shared__ alignas(32) bf16_t Bs[4 * 512];
  const int tid = threadIdx.x, lane = tid & 31, wv = tid >> 5;
  const int m0g = blockIdx.x * 128, n0g = blockIdx.y * 64;
  v8f acc[4] = {};
  const int kchunks = (K + 31) >> 5;
  for (int kc = 0; kc < kchunks; ++kc) {
    for (int s = tid; s < 512; s += 256) {
      int grp = s & 1, ln = (s >> 1) & 31, mtile = s >> 6;
      int m = ln & 15, hi = ln >> 4;
      int gr = m0g + mtile * 16 + m;
      int kb = kc * 32 + hi * 16 + grp * 8;
      u32x4 d;
      if (a_bf16 && gr < M && kb + 8 <= K) {
        d = *(const u32x4*)((const bf16_t*)Aptr + (size_t)gr * K + kb);
      } else {
        float v[8];
#pragma unroll
        for (int e = 0; e < 8; ++e) {
          int gk = kb + e;
          float f = 0.f;
          if (gr < M && gk < K)
            f = a_bf16 ? bf2float(((const bf16_t*)Aptr)[(size_t)gr * K + gk])
                       : ((const float*)Aptr)[(size_t)gr * K + gk];
          v[e] = f;
        }
        d[0] = pack2bf(v[0], v[1]); d[1] = pack2bf(v[2], v[3]);
        d[2] = pack2bf(v[4], v[5]); d[3] = pack2bf(v[6], v[7]);
      }
      *(u32x4*)(As + (size_t)mtile * 512 + ln * 16 + grp * 8) = d;
    }
    for (int s = tid; s < 128; s += 256) {
      int ln = s & 31, ntile = s >> 5;
      int gn = n0g + ntile * 16 + (ln & 15);
      int kb = kc * 32 + ((ln >> 4) << 4);
      unsigned pk[8];
#pragma unroll
      for (int p = 0; p < 8; ++p) {
        int k0 = kb + 2 * p;
        float f0 = (k0 < K && gn < N) ? Bw[(size_t)k0 * N + gn] : 0.f;
        float f1 = (k0 + 1 < K && gn < N) ? Bw[(size_t)(k0 + 1) * N + gn] : 0.f;
        pk[p] = pack2bf(f0, f1);
      }
      u32x4* dst = (u32x4*)(Bs + (size_t)ntile * 512 + ln * 16);
      u32x4 lo = { pk[0], pk[1], pk[2], pk[3] };
      u32x4 hi4 = { pk[4], pk[5], pk[6], pk[7] };
      dst[0] = lo; dst[1] = hi4;
    }
    __syncthreads();
    v16bf af = load_frag(As + (size_t)wv * 512, lane);
#pragma unroll
    for (int t = 0; t < 4; ++t) {
      v16bf bfv = load_frag(Bs + (size_t)t * 512, lane);
      acc[t] = wmma_bf16(af, bfv, acc[t]);
    }
    __syncthreads();
  }
  const int half8 = (lane >> 4) * 8, nlo = lane & 15;
#pragma unroll
  for (int t = 0; t < 4; ++t)
#pragma unroll
    for (int r = 0; r < 8; ++r) {
      int m = m0g + wv * 16 + half8 + r, n = n0g + t * 16 + nlo;
      if (m < M && n < N) {
        float v = acc[t][r];
        if (bias)   v += bias[n];
        if (addend) v += addend[(size_t)m * N + n];
        if (act == 1) v = fmaxf(v, 0.f);
        if (outF) outF[(size_t)m * N + n] = v;
        if (outB) outB[(size_t)m * N + n] = f2bf(v);
      }
    }
}

__global__ void zero_f32(float* p, int n) {
  int i = blockIdx.x * blockDim.x + threadIdx.x;
  if (i < n) p[i] = 0.f;
}

// ---------------------------------------------------------------------------
extern "C" void kernel_launch(void* const* d_in, const int* in_sizes, int n_in,
                              void* d_out, int out_size, void* d_ws, size_t ws_size,
                              hipStream_t stream)
{
  (void)in_sizes; (void)n_in; (void)out_size; (void)ws_size;
  const float* h     = (const float*)d_in[0];
  const int*   adj   = (const int*)d_in[1];
  const float* edge  = (const float*)d_in[2];
  const float* fp    = (const float*)d_in[3];
  const float* Whead = (const float*)d_in[4];
  const float* a1h   = (const float*)d_in[5];
  const float* a2h   = (const float*)d_in[6];
  const float* aeh   = (const float*)d_in[7];
  const float* Wout  = (const float*)d_in[8];
  const float* a1o   = (const float*)d_in[9];
  const float* a2o   = (const float*)d_in[10];
  const float* aeo   = (const float*)d_in[11];
  const float* fc1w  = (const float*)d_in[12];
  const float* fc1b  = (const float*)d_in[13];
  const float* fc2w  = (const float*)d_in[14];
  const float* fc2b  = (const float*)d_in[15];
  const float* qw    = (const float*)d_in[16];
  const float* qb    = (const float*)d_in[17];
  // d_in[18]/d_in[19] (k_w, k_b) are mathematically dead: softmax over a
  // size-1 axis makes the fusion weight identically 1, so fused = q + v.
  const float* vw    = (const float*)d_in[20];
  const float* vb    = (const float*)d_in[21];
  const float* ow    = (const float*)d_in[22];
  const float* ob    = (const float*)d_in[23];
  const float* f1w   = (const float*)d_in[24];
  const float* f1b   = (const float*)d_in[25];
  const float* f2w   = (const float*)d_in[26];
  const float* f2b   = (const float*)d_in[27];
  float* out = (float*)d_out;

  char* ws = (char*)d_ws;
  bf16_t* xB   = (bf16_t*)(ws + 0);               //  64 MB  x (bf16)  [B*N, 512]
  bf16_t* WxB  = (bf16_t*)(ws + 67108864ull);     //  64 MB  Wx (bf16) [B*N, 512]
  float*  t1   = (float*)(ws + 134217728ull);     // 256 KB
  float*  t2   = (float*)(ws + 134479872ull);     // 256 KB (contiguous after t1)
  float*  gat  = (float*)(ws + 134742016ull);     //   1 MB  gat_out f32 [B,512]
  bf16_t* fpnh = (bf16_t*)(ws + 135790592ull);    // 512 KB  relu(fp@fc1+b) bf16
  bf16_t* fpnB = (bf16_t*)(ws + 136314880ull);    // 512 KB  fpn_out bf16
  float*  qf   = (float*)(ws + 136839168ull);     //   1 MB  q f32
  bf16_t* fpre = (bf16_t*)(ws + 137887744ull);    // 512 KB  q+v bf16
  bf16_t* fusd = (bf16_t*)(ws + 138412032ull);    // 512 KB  fused bf16
  bf16_t* hid  = (bf16_t*)(ws + 138936320ull);    // 512 KB  ffn hidden bf16

  gat1_fused<<<dim3(8, 512), 256, 0, stream>>>(h, adj, edge, Whead, a1h, a2h, aeh, xB);
  zero_f32<<<512, 256, 0, stream>>>(t1, 131072);
  gemm_wout<<<dim3(512, 8), 256, 24576, stream>>>(xB, Wout, a1o, a2o, WxB, t1, t2);
  gat2_fused<<<512, 256, 0, stream>>>(adj, edge, aeo, t1, t2, WxB, gat);
  gemm_generic<<<dim3(4, 8), 256, 0, stream>>>(fp, 0, fc1w, fc1b, nullptr,
                                               nullptr, fpnh, 512, 512, 1489, 1);
  gemm_generic<<<dim3(4, 8), 256, 0, stream>>>(fpnh, 1, fc2w, fc2b, nullptr,
                                               nullptr, fpnB, 512, 512, 512, 0);
  gemm_generic<<<dim3(4, 8), 256, 0, stream>>>(gat, 0, qw, qb, nullptr,
                                               qf, nullptr, 512, 512, 512, 0);
  gemm_generic<<<dim3(4, 8), 256, 0, stream>>>(fpnB, 1, vw, vb, qf,
                                               nullptr, fpre, 512, 512, 512, 0);
  gemm_generic<<<dim3(4, 8), 256, 0, stream>>>(fpre, 1, ow, ob, nullptr,
                                               nullptr, fusd, 512, 512, 512, 1);
  gemm_generic<<<dim3(4, 8), 256, 0, stream>>>(fusd, 1, f1w, f1b, nullptr,
                                               nullptr, hid, 512, 512, 512, 1);
  gemm_generic<<<dim3(4, 1), 256, 0, stream>>>(hid, 1, f2w, f2b, nullptr,
                                               out, nullptr, 512, 12, 512, 0);
}
